// NeuralBlendFieldCVPR_76433237999818
// MI455X (gfx1250) — matrix operand order repaired and checked by reading, not compiled, not run on hardware
//
#include <hip/hip_runtime.h>
#include <hip/hip_bf16.h>

#define K_      5
#define B_      8
#define N_      16384
#define H_      128
#define HH_     256
#define LCODE_  128
#define PE_DIM_ 39
#define P_TOT_  54785
#define OMEGA_  30.0f

// f16 weight image per (b,k), matching the LDS layout exactly (halves):
//   Wt0 [128 cols][64 feats]  @ 0      (posenc layer, padded 39->64)
//   Wt1 [128][128]            @ 8192
//   Wt2 [128][128]            @ 24576
//   Wt3 [128][128]            @ 40960   -> total 57344 halves = 114688 B
#define WIMG_HALVES 57344
// f32 side vector per (b,k): b0@0 b1@128 b2@256 b3@384 Wo@512 bo@640 (641, stride 656)
#define BVEC_STRIDE 656

typedef __attribute__((ext_vector_type(16))) _Float16 v16h;
typedef __attribute__((ext_vector_type(8)))  _Float16 v8h;
typedef __attribute__((ext_vector_type(8)))  float    v8f;

union AFrag { v16h v; v8h h[2]; };
union HPack { v8h v; _Float16 e[8]; };

// A-matrix 16x32 f16 fragment (ISA 7.12.2): lane L<16: row L, K={0..7,16..23};
// lane L+16: row L, K={8..15,24..31}. Abuf row-major, stride 128 halves.
__device__ __forceinline__ v16h load_a_frag(const _Float16* Abuf, int lane, int kbase) {
  const int row = lane & 15, hi = lane >> 4;
  AFrag f;
  f.h[0] = *(const v8h*)(Abuf + row * 128 + kbase + 8 * hi);
  f.h[1] = *(const v8h*)(Abuf + row * 128 + kbase + 16 + 8 * hi);
  return f.v;
}

// B-matrix 32x16 f16 fragment: lane L: column N=L%16, K={8h..8h+7, 16+8h..16+8h+7}, h=L/16.
// Weights transposed in LDS: Wt[j][f], f contiguous, stride fstride.
__device__ __forceinline__ v16h load_b_frag(const _Float16* Wt, int lane, int jbase,
                                            int kbase, int fstride) {
  const int col = lane & 15, hi = lane >> 4;
  const _Float16* s = Wt + (size_t)(jbase + col) * fstride + kbase + 8 * hi;
  AFrag f;
  f.h[0] = *(const v8h*)(s);
  f.h[1] = *(const v8h*)(s + 16);
  return f.v;
}

// posenc feature value; f becomes compile-time constant after unrolling.
__device__ __forceinline__ float pe_val(int f, float px, float py, float pz) {
  if (f < 3) return (f == 0) ? px : ((f == 1) ? py : pz);
  if (f >= PE_DIM_) return 0.f;
  const int q = f - 3, l = q / 6, r = q - 6 * l;
  const int d = (r >= 3) ? r - 3 : r;
  const float xd = (d == 0) ? px : ((d == 1) ? py : pz);
  const float arg = xd * (3.14159265358979f * (float)(1 << l));
  return (r >= 3) ? __cosf(arg) : __sinf(arg);
}

// ---------------------------------------------------------------------------
// Kernel 1a: hypernet trunk z2 = relu(relu(code@hW1+hb1)@hW2+hb2); also
// zero-fills the 39->64 pad region of the f16 weight image.  (40 blocks)
// ---------------------------------------------------------------------------
__global__ __launch_bounds__(256)
void hyper_z_kernel(const float* __restrict__ code,
                    const float* __restrict__ hW1, const float* __restrict__ hb1,
                    const float* __restrict__ hW2, const float* __restrict__ hb2,
                    float* __restrict__ z2buf, _Float16* __restrict__ whalf) {
  __shared__ float cs[LCODE_];
  __shared__ float z1[HH_];
  const int bk = blockIdx.x;
  const int b = bk / K_, k = bk % K_;
  const int tid = threadIdx.x;
  if (tid < LCODE_) cs[tid] = code[(size_t)(b * K_ + k) * LCODE_ + tid];
  __syncthreads();
  float a = hb1[k * HH_ + tid];
  const float* w1 = hW1 + (size_t)k * LCODE_ * HH_;
  for (int d = 0; d < LCODE_; ++d) a += cs[d] * w1[(size_t)d * HH_ + tid];
  z1[tid] = fmaxf(a, 0.f);
  __syncthreads();
  float a2 = hb2[k * HH_ + tid];
  const float* w2 = hW2 + (size_t)k * HH_ * HH_;
  for (int d = 0; d < HH_; ++d) a2 += z1[d] * w2[(size_t)d * HH_ + tid];
  z2buf[(size_t)bk * HH_ + tid] = fmaxf(a2, 0.f);
  _Float16* w0 = whalf + (size_t)bk * WIMG_HALVES;
  for (int i = tid; i < 128 * 64; i += 256)
    if ((i & 63) >= PE_DIM_) w0[i] = (_Float16)0.f;
}

// ---------------------------------------------------------------------------
// Kernel 1b: head GEMM; routes each parameter straight into the f16
// transposed weight image + f32 bias vector. grid (chunks, K).
// ---------------------------------------------------------------------------
__global__ __launch_bounds__(256)
void hyper_head_kernel(const float* __restrict__ z2buf,
                       const float* __restrict__ headW, const float* __restrict__ headb,
                       _Float16* __restrict__ whalf, float* __restrict__ bvec) {
  __shared__ float z[B_][HH_];
  const int k = blockIdx.y;
  const int tid = threadIdx.x;
  for (int i = tid; i < B_ * HH_; i += 256) {
    int b = i >> 8, d = i & 255;
    z[b][d] = z2buf[(size_t)(b * K_ + k) * HH_ + d];
  }
  __syncthreads();
  const int j = blockIdx.x * 256 + tid;
  if (j >= P_TOT_) return;
  float acc[B_];
#pragma unroll
  for (int b = 0; b < B_; ++b) acc[b] = 0.f;
  const float* w = headW + (size_t)k * HH_ * P_TOT_ + j;
  for (int d = 0; d < HH_; ++d) {
    if ((d & 15) == 0)
      __builtin_prefetch((const void*)(w + (size_t)(d + 32) * P_TOT_), 0, 0);
    float wv = w[(size_t)d * P_TOT_];
#pragma unroll
    for (int b = 0; b < B_; ++b) acc[b] += z[b][d] * wv;
  }
  const float hb = headb[(size_t)k * P_TOT_ + j];
#pragma unroll
  for (int b = 0; b < B_; ++b) {
    const float val = acc[b] + hb;
    const size_t bko = (size_t)(b * K_ + k);
    _Float16* wi = whalf + bko * WIMG_HALVES;
    float* bv = bvec + bko * BVEC_STRIDE;
    if (j < 4992) {                       // W0[f][c] -> Wt0[c*64+f]
      int f = j >> 7, c = j & 127;
      wi[c * 64 + f] = (_Float16)val;
    } else if (j < 5120) {                // b0
      bv[j - 4992] = val;
    } else if (j < 54656) {               // hidden layers
      int q = j - 5120;
      int l = q / 16512, r = q - l * 16512;
      if (r < 16384) {                    // Wl[f][c] -> Wtl[c*128+f]
        int f = r >> 7, c = r & 127;
        wi[8192 + l * 16384 + c * 128 + f] = (_Float16)val;
      } else {
        bv[128 + l * 128 + (r - 16384)] = val;
      }
    } else if (j < 54784) {               // Wo
      bv[512 + (j - 54656)] = val;
    } else {                              // bo
      bv[640] = val;
    }
  }
}

// ---------------------------------------------------------------------------
// Kernel 2: SIREN field. grid (32, B*K), 256 threads = 8 INDEPENDENT waves.
// Weights async-copied into LDS once; each wave owns TWO private 16-point
// tiles (B-fragments reused for 2 WMMAs) with ping-pong activation buffers;
// A-fragments hoisted into registers; no workgroup barriers in the hot loop.
// ---------------------------------------------------------------------------
__global__ __launch_bounds__(256)
void siren_field_kernel(const _Float16* __restrict__ whalf, const float* __restrict__ bvec,
                        const float* __restrict__ keypoints, const float* __restrict__ xyz,
                        float* __restrict__ outs) {
  extern __shared__ char smem[];
  _Float16* Wt0 = (_Float16*)smem;            // @0      (128x64)
  _Float16* Wt1 = Wt0 + 8192;                 // @16384B (128x128)
  _Float16* Wt2 = Wt0 + 24576;                // @49152B
  _Float16* Wt3 = Wt0 + 40960;                // @81920B ; weights end @114688B
  float* biasL = (float*)(smem + 245760);     // 4*128 f32
  float* WoL   = biasL + 512;                 // 128 f32
  float* scal  = WoL + 128;                   // [0]=bo

  const int tid  = threadIdx.x;
  const int lane = tid & 31;
  const int wv   = tid >> 5;
  const int bk   = blockIdx.y;
  const int b    = bk / K_;

  // per-wave activation buffers: 2 tiles x ping/pong x 2048 halves
  _Float16* Abase = Wt0 + 57344 + wv * 8192;
  _Float16* P0t0 = Abase;            // ping tile0
  _Float16* P1t0 = Abase + 2048;     // pong tile0
  _Float16* P0t1 = Abase + 4096;     // ping tile1
  _Float16* P1t1 = Abase + 6144;     // pong tile1

  // ---- async-copy the 114688-byte f16 weight image into LDS ----
  {
    const unsigned long long gbase =
        (unsigned long long)(const void*)(whalf + (size_t)bk * WIMG_HALVES);
    const unsigned lbase = (unsigned)(size_t)(void*)Wt0;
    for (int i = tid; i < 7168; i += 256) {          // 7168 x 16B
      const unsigned off = (unsigned)i * 16u;
      asm volatile("global_load_async_to_lds_b128 %0, %1, %2"
                   :: "v"(lbase + off), "v"(off), "s"(gbase) : "memory");
    }
    asm volatile("s_wait_asynccnt 0x0" ::: "memory");
  }
  const float* bv = bvec + (size_t)bk * BVEC_STRIDE;
  if (tid < 128) {
    biasL[tid]       = bv[tid];
    biasL[128 + tid] = bv[128 + tid];
    biasL[256 + tid] = bv[256 + tid];
    biasL[384 + tid] = bv[384 + tid];
    WoL[tid]         = bv[512 + tid];
  }
  if (tid == 0) scal[0] = bv[640];
  __syncthreads();

  const float bo = scal[0];
  const float kx = keypoints[(size_t)bk * 3 + 0];
  const float ky = keypoints[(size_t)bk * 3 + 1];
  const float kz = keypoints[(size_t)bk * 3 + 2];

  const int colD = lane & 15, hiD = lane >> 4;
  const int ntiles = N_ / 32;                 // 32 points per wave-iteration
  const int wstride = gridDim.x * 8;
  for (int t = blockIdx.x * 8 + wv; t < ntiles; t += wstride) {
    const int n0 = t * 32;
    // ---- posenc for both tiles: lane = (row m, feature half hi) ----
    {
      const int m = lane & 15, hi = lane >> 4;
#pragma unroll
      for (int u = 0; u < 2; ++u) {
        _Float16* A = u ? P0t1 : P0t0;
        const size_t base3 = ((size_t)b * N_ + n0 + u * 16 + m) * 3;
        const float px = xyz[base3 + 0] - kx;
        const float py = xyz[base3 + 1] - ky;
        const float pz = xyz[base3 + 2] - kz;
        if (hi == 0) {
#pragma unroll
          for (int g = 0; g < 4; ++g) {
            HPack pk;
#pragma unroll
            for (int e = 0; e < 8; ++e) pk.e[e] = (_Float16)pe_val(g * 8 + e, px, py, pz);
            *(v8h*)(A + m * 128 + g * 8) = pk.v;
          }
        } else {
#pragma unroll
          for (int g = 0; g < 4; ++g) {
            HPack pk;
#pragma unroll
            for (int e = 0; e < 8; ++e) pk.e[e] = (_Float16)pe_val(32 + g * 8 + e, px, py, pz);
            *(v8h*)(A + m * 128 + 32 + g * 8) = pk.v;
          }
        }
      }
    }
    asm volatile("s_wait_dscnt 0x0" ::: "memory");

    // ---- layer 0: [16x64] @ [64x128], A-frags hoisted, B reused x2 ----
    {
      v16h a00 = load_a_frag(P0t0, lane, 0);
      v16h a01 = load_a_frag(P0t0, lane, 32);
      v16h a10 = load_a_frag(P0t1, lane, 0);
      v16h a11 = load_a_frag(P0t1, lane, 32);
#pragma unroll
      for (int jt = 0; jt < 8; ++jt) {
        v8f acc0 = {}, acc1 = {};
        {
          v16h bm = load_b_frag(Wt0, lane, jt * 16, 0, 64);
          acc0 = __builtin_amdgcn_wmma_f32_16x16x32_f16(false, a00, false, bm, (short)0, acc0, false, false);
          acc1 = __builtin_amdgcn_wmma_f32_16x16x32_f16(false, a10, false, bm, (short)0, acc1, false, false);
        }
        {
          v16h bm = load_b_frag(Wt0, lane, jt * 16, 32, 64);
          acc0 = __builtin_amdgcn_wmma_f32_16x16x32_f16(false, a01, false, bm, (short)0, acc0, false, false);
          acc1 = __builtin_amdgcn_wmma_f32_16x16x32_f16(false, a11, false, bm, (short)0, acc1, false, false);
        }
        const float bj = biasL[jt * 16 + colD];
#pragma unroll
        for (int r = 0; r < 8; ++r) {
          const int o = (r + 8 * hiD) * 128 + jt * 16 + colD;
          P1t0[o] = (_Float16)__sinf(OMEGA_ * (acc0[r] + bj));
          P1t1[o] = (_Float16)__sinf(OMEGA_ * (acc1[r] + bj));
        }
      }
    }
    asm volatile("s_wait_dscnt 0x0" ::: "memory");

    // ---- hidden layers 1..3: [16x128] @ [128x128], ping-pong ----
    _Float16* Ain0 = P1t0;  _Float16* Ain1 = P1t1;
    _Float16* Aout0 = P0t0; _Float16* Aout1 = P0t1;
    _Float16* Wts[3] = {Wt1, Wt2, Wt3};
#pragma unroll
    for (int l = 0; l < 3; ++l) {
      const float* bl = biasL + (l + 1) * 128;
      v16h a0[4], a1[4];
#pragma unroll
      for (int ks = 0; ks < 4; ++ks) {
        a0[ks] = load_a_frag(Ain0, lane, ks * 32);
        a1[ks] = load_a_frag(Ain1, lane, ks * 32);
      }
#pragma unroll
      for (int jt = 0; jt < 8; ++jt) {
        v8f acc0 = {}, acc1 = {};
#pragma unroll
        for (int ks = 0; ks < 4; ++ks) {
          v16h bm = load_b_frag(Wts[l], lane, jt * 16, ks * 32, 128);
          acc0 = __builtin_amdgcn_wmma_f32_16x16x32_f16(false, a0[ks], false, bm, (short)0, acc0, false, false);
          acc1 = __builtin_amdgcn_wmma_f32_16x16x32_f16(false, a1[ks], false, bm, (short)0, acc1, false, false);
        }
        const float bj = bl[jt * 16 + colD];
#pragma unroll
        for (int r = 0; r < 8; ++r) {
          const int o = (r + 8 * hiD) * 128 + jt * 16 + colD;
          Aout0[o] = (_Float16)__sinf(OMEGA_ * (acc0[r] + bj));
          Aout1[o] = (_Float16)__sinf(OMEGA_ * (acc1[r] + bj));
        }
      }
      asm volatile("s_wait_dscnt 0x0" ::: "memory");
      _Float16* tmp;
      tmp = Ain0; Ain0 = Aout0; Aout0 = tmp;
      tmp = Ain1; Ain1 = Aout1; Aout1 = tmp;
    }

    // ---- output 128->1: one full row-dot per lane (32 rows total) ----
    {
      const _Float16* hr = (lane < 16 ? Ain0 : Ain1) + (lane & 15) * 128;
      float s = bo;
#pragma unroll
      for (int c = 0; c < 16; ++c) {
        v8h hv = *(const v8h*)(hr + c * 8);
#pragma unroll
        for (int e = 0; e < 8; ++e) s += (float)hv[e] * WoL[c * 8 + e];
      }
      outs[(size_t)bk * N_ + n0 + lane] = s;
    }
    asm volatile("s_wait_dscnt 0x0" ::: "memory");
  }
}

// ---------------------------------------------------------------------------
// Kernel 3: weight net (3->128->128->5, layernorm + leaky_relu + softmax)
// ---------------------------------------------------------------------------
__global__ __launch_bounds__(128)
void weightnet_kernel(const float* __restrict__ xyz,
                      const float* __restrict__ fW1, const float* __restrict__ fb1,
                      const float* __restrict__ fg1, const float* __restrict__ fB1,
                      const float* __restrict__ fW2, const float* __restrict__ fb2,
                      const float* __restrict__ fg2, const float* __restrict__ fB2,
                      const float* __restrict__ fW3, const float* __restrict__ fb3,
                      const float* __restrict__ fg3, const float* __restrict__ fB3,
                      float* __restrict__ wout) {
  extern __shared__ char smem[];
  float* W2s = (float*)smem;        // 128*128
  float* W1s = W2s + 128 * 128;     // 3*128
  float* W3s = W1s + 3 * 128;       // 128*5
  float* vec = W3s + 128 * 5;       // b1,g1,B1,b2,g2,B2 : 6*128
  float* v3  = vec + 6 * 128;       // b3,g3,B3 : 15 (+1 pad)
  float* h1  = v3 + 16;             // 128
  float* h2  = h1 + 128;            // 128
  float* red = h2 + 128;            // 8
  float* lgs = red + 8;             // 5

  const int tid = threadIdx.x;
  const int wv = tid >> 5;
  for (int i = tid; i < 128 * 128; i += 128) W2s[i] = fW2[i];
  for (int i = tid; i < 3 * 128; i += 128)  W1s[i] = fW1[i];
  for (int i = tid; i < 128 * 5; i += 128)  W3s[i] = fW3[i];
  vec[tid] = fb1[tid]; vec[128 + tid] = fg1[tid]; vec[256 + tid] = fB1[tid];
  vec[384 + tid] = fb2[tid]; vec[512 + tid] = fg2[tid]; vec[640 + tid] = fB2[tid];
  if (tid < 5) { v3[tid] = fb3[tid]; v3[5 + tid] = fg3[tid]; v3[10 + tid] = fB3[tid]; }
  __syncthreads();

  for (int pidx = blockIdx.x; pidx < B_ * N_; pidx += gridDim.x) {
    const float x0 = xyz[(size_t)pidx * 3];
    const float x1 = xyz[(size_t)pidx * 3 + 1];
    const float x2 = xyz[(size_t)pidx * 3 + 2];
    float c = x0 * W1s[tid] + x1 * W1s[128 + tid] + x2 * W1s[256 + tid] + vec[tid];
    float a = c, s2 = c * c;
#pragma unroll
    for (int off = 16; off; off >>= 1) { a += __shfl_xor(a, off, 32); s2 += __shfl_xor(s2, off, 32); }
    if ((tid & 31) == 0) { red[wv * 2] = a; red[wv * 2 + 1] = s2; }
    __syncthreads();
    float sum = red[0] + red[2] + red[4] + red[6];
    float sq  = red[1] + red[3] + red[5] + red[7];
    __syncthreads();
    float mu = sum * (1.f / 128.f), var = sq * (1.f / 128.f) - mu * mu;
    float tv = (c - mu) * rsqrtf(var + 1e-5f) * vec[128 + tid] + vec[256 + tid];
    tv = (tv > 0.f) ? tv : 0.01f * tv;
    h1[tid] = tv;
    __syncthreads();
    float c2 = vec[384 + tid];
    for (int d = 0; d < 128; ++d) c2 += h1[d] * W2s[d * 128 + tid];
    a = c2; s2 = c2 * c2;
#pragma unroll
    for (int off = 16; off; off >>= 1) { a += __shfl_xor(a, off, 32); s2 += __shfl_xor(s2, off, 32); }
    if ((tid & 31) == 0) { red[wv * 2] = a; red[wv * 2 + 1] = s2; }
    __syncthreads();
    sum = red[0] + red[2] + red[4] + red[6];
    sq  = red[1] + red[3] + red[5] + red[7];
    __syncthreads();
    mu = sum * (1.f / 128.f); var = sq * (1.f / 128.f) - mu * mu;
    tv = (c2 - mu) * rsqrtf(var + 1e-5f) * vec[512 + tid] + vec[640 + tid];
    tv = (tv > 0.f) ? tv : 0.01f * tv;
    h2[tid] = tv;
    __syncthreads();
    if (tid < K_) {
      float lg = v3[tid];
      for (int d = 0; d < 128; ++d) lg += h2[d] * W3s[d * K_ + tid];
      lgs[tid] = lg;
    }
    __syncthreads();
    if (tid == 0) {
      float m5 = 0.f, q5 = 0.f;
      for (int j = 0; j < K_; ++j) { m5 += lgs[j]; q5 += lgs[j] * lgs[j]; }
      m5 *= 0.2f; q5 = q5 * 0.2f - m5 * m5;
      const float inv = rsqrtf(q5 + 1e-5f);
      float ln[K_]; float mx = -1e30f;
      for (int j = 0; j < K_; ++j) {
        ln[j] = (lgs[j] - m5) * inv * v3[5 + j] + v3[10 + j];
        mx = fmaxf(mx, ln[j]);
      }
      float se = 0.f;
      for (int j = 0; j < K_; ++j) { ln[j] = __expf(ln[j] - mx); se += ln[j]; }
      const float rs = 1.f / se;
      for (int j = 0; j < K_; ++j) wout[(size_t)pidx * K_ + j] = ln[j] * rs;
    }
    __syncthreads();
  }
}

// ---------------------------------------------------------------------------
// Kernel 4: blend  out[b,n] = sum_k weights[b,n,k] * outs[b,k,n]
// ---------------------------------------------------------------------------
__global__ __launch_bounds__(256)
void combine_kernel(const float* __restrict__ wts, const float* __restrict__ outs,
                    float* __restrict__ out) {
  const int p = blockIdx.x * blockDim.x + threadIdx.x;
  if (p >= B_ * N_) return;
  const int b = p >> 14;
  const int n = p & (N_ - 1);
  float acc = 0.f;
#pragma unroll
  for (int k = 0; k < K_; ++k)
    acc += wts[(size_t)p * K_ + k] * outs[((size_t)(b * K_ + k) << 14) + n];
  out[p] = acc;
}

// ---------------------------------------------------------------------------
extern "C" void kernel_launch(void* const* d_in, const int* in_sizes, int n_in,
                              void* d_out, int out_size, void* d_ws, size_t ws_size,
                              hipStream_t stream) {
  const float* keypoints = (const float*)d_in[0];
  const float* xyz   = (const float*)d_in[1];
  const float* code  = (const float*)d_in[2];
  const float* hW1   = (const float*)d_in[3];
  const float* hb1   = (const float*)d_in[4];
  const float* hW2   = (const float*)d_in[5];
  const float* hb2   = (const float*)d_in[6];
  const float* headW = (const float*)d_in[7];
  const float* headb = (const float*)d_in[8];
  const float* fW1 = (const float*)d_in[9];
  const float* fb1 = (const float*)d_in[10];
  const float* fg1 = (const float*)d_in[11];
  const float* fB1 = (const float*)d_in[12];
  const float* fW2 = (const float*)d_in[13];
  const float* fb2 = (const float*)d_in[14];
  const float* fg2 = (const float*)d_in[15];
  const float* fB2 = (const float*)d_in[16];
  const float* fW3 = (const float*)d_in[17];
  const float* fb3 = (const float*)d_in[18];
  const float* fg3 = (const float*)d_in[19];
  const float* fB3 = (const float*)d_in[20];

  float* out = (float*)d_out;
  float* wts = out + (size_t)B_ * N_;   // weights region of d_out (B*N*K)

  _Float16* whalf = (_Float16*)d_ws;                                  // 40*57344 halves
  float* bvec  = (float*)(whalf + (size_t)B_ * K_ * WIMG_HALVES);     // 40*656
  float* z2buf = bvec + (size_t)B_ * K_ * BVEC_STRIDE;                // 40*256
  float* outs  = z2buf + (size_t)B_ * K_ * HH_;                       // 40*16384

  hyper_z_kernel<<<B_ * K_, 256, 0, stream>>>(code, hW1, hb1, hW2, hb2, z2buf, whalf);

  const int head_chunks = (P_TOT_ + 255) / 256;
  hyper_head_kernel<<<dim3(head_chunks, K_), 256, 0, stream>>>(z2buf, headW, headb,
                                                               whalf, bvec);

  // LDS: weights 114688B + 8 waves * 4 * 4096B A-buffers + (512+128+4)*4B
  const size_t siren_lds = 114688 + 131072 + (512 + 128 + 4) * 4;
  siren_field_kernel<<<dim3(32, B_ * K_), 256, siren_lds, stream>>>(whalf, bvec,
                                                                    keypoints, xyz, outs);

  const size_t wn_lds = (size_t)(128 * 128 + 3 * 128 + 128 * 5 + 6 * 128 + 16
                                 + 128 + 128 + 8 + 5 + 3) * 4;
  weightnet_kernel<<<512, 128, wn_lds, stream>>>(xyz, fW1, fb1, fg1, fB1,
                                                 fW2, fb2, fg2, fB2,
                                                 fW3, fb3, fg3, fB3, wts);

  combine_kernel<<<(B_ * N_ + 255) / 256, 256, 0, stream>>>(wts, outs, out);
}
